// FlowGuidedDCNAlign_84439057039453
// MI455X (gfx1250) — matrix operand order repaired and checked by reading, not compiled, not run on hardware
//
#include <hip/hip_runtime.h>
#include <hip/hip_bf16.h>
#include <math.h>

// ---------------------------------------------------------------------------
// FlowGuidedDCNAlign on MI455X (gfx1250, wave32, WMMA).
// All GEMM-like convolutions run on v_wmma_f32_16x16x32_f16 (f16 in, f32 acc).
// Block tile 64x64, 8 waves, 2-way M register blocking per wave:
//   A-tile reused 4x, B-tile 2x -> 32 MACs per LDS-staged element.
// Workspace arena requirement: ~264 MB (see offsets in kernel_launch).
// ---------------------------------------------------------------------------

typedef __attribute__((ext_vector_type(16))) _Float16 v16h;
typedef __attribute__((ext_vector_type(8)))  float    v8f;

#define ACT_NONE  0
#define ACT_RELU  1
#define ACT_LRELU 2

// WMMA fragment layouts (ISA 7.12.2, 16-bit, wave32):
//   A (16x32): lane L -> M = L&15;  element e -> K = e + 8*(e>=8) + 8*(L>=16)
//   B (32x16): lane L -> N = L&15;  element e -> K = e + 8*(e>=8) + 8*(L>=16)
//   C (16x16 f32): element r -> M = r + 8*(L>=16), N = L&15
__device__ __forceinline__ int kmapA(int e, int L) {
    return (e & 7) + ((L & 16) ? 8 : 0) + ((e & 8) ? 16 : 0);
}
__device__ __forceinline__ int kmapB(int e, int L) {
    return (e & 7) + ((L & 16) ? 8 : 0) + ((e & 8) ? 16 : 0);
}

// ---------------------------------------------------------------------------
// Generic conv-as-GEMM with on-the-fly im2col.
//   Y[img][m][p] = act( sum_k W[m][k] * im2col(X[img])[k][p] + bias[m] (+resid) )
// W must be a power of two (wshift = log2 W).
// grid = (P/64, ceil(M/64), images), block = 256.
// ---------------------------------------------------------------------------
__global__ __launch_bounds__(256)
void conv_wmma(const float* __restrict__ X, const float* __restrict__ Wt,
               const float* __restrict__ bias, const float* __restrict__ resid,
               float* __restrict__ Y, int M, int Cin, int ksq, int pad,
               int H, int wshift, int act)
{
    __shared__ alignas(32) _Float16 As[4][512];   // 64 rows of M, frag-ordered
    __shared__ alignas(32) _Float16 Bs[4][512];   // 64 cols of N, frag-ordered

    const int W      = 1 << wshift;
    const int P      = H << wshift;
    const int K      = Cin * ksq;
    const int img    = blockIdx.z;
    const float* Xi  = X + (size_t)img * Cin * P;
    const int pbase  = blockIdx.x * 64;
    const int Mblock = blockIdx.y * 64;
    const int tid    = threadIdx.x;
    const int lane   = tid & 31;
    const int wave   = tid >> 5;
    const int wm     = wave >> 2;      // 0..1 : which 32-row half of the A tile
    const int wn     = wave & 3;       // 0..3 : which 16-col slice of the B tile

    v8f acc0 = {};
    v8f acc1 = {};
    const int nk = (K + 31) >> 5;
    for (int kc = 0; kc < nk; ++kc) {
        const int k0 = kc << 5;
        __syncthreads();               // previous fragment reads complete
        #pragma unroll
        for (int j = 0; j < 8; ++j) {
            const int g    = tid * 8 + j;        // 0..2047
            const int frag = g >> 9;             // 0..3
            const int L    = (g >> 4) & 31;
            const int e    = g & 15;
            // ---- A tile (weights) in fragment order ----
            {
                const int m = Mblock + frag * 16 + (L & 15);
                const int k = k0 + kmapA(e, L);
                const float v = (m < M && k < K) ? Wt[(size_t)m * K + k] : 0.f;
                As[frag][(L << 4) + e] = (_Float16)v;
            }
            // ---- B tile (im2col of X) in fragment order ----
            {
                const int n = frag * 16 + (L & 15);
                const int k = k0 + kmapB(e, L);
                float v = 0.f;
                if (k < K) {
                    const int p = pbase + n;
                    const int y = p >> wshift, x = p & (W - 1);
                    int c, sy, sx;
                    if (ksq == 1) { c = k; sy = y; sx = x; }
                    else {
                        c = k / 9;                 // constant divisor -> mul/shift
                        const int t  = k - c * 9;
                        const int ky = t / 3;
                        sy = y + ky - pad;
                        sx = x + (t - ky * 3) - pad;
                    }
                    if (sy >= 0 && sy < H && sx >= 0 && sx < W)
                        v = Xi[((size_t)c * H + sy) << wshift | sx];
                }
                Bs[frag][(L << 4) + e] = (_Float16)v;
            }
        }
        // speculative prefetch of next K-chunk of the weight stream
        if (kc + 1 < nk) {
            const int pm = Mblock + (tid & 63);
            __builtin_prefetch(Wt + (size_t)pm * K + k0 + 32, 0, 0);
        }
        __syncthreads();               // staging visible
        v16h a0 = *reinterpret_cast<const v16h*>(&As[wm * 2 + 0][lane << 4]);
        v16h a1 = *reinterpret_cast<const v16h*>(&As[wm * 2 + 1][lane << 4]);
        v16h b  = *reinterpret_cast<const v16h*>(&Bs[wn][lane << 4]);
        acc0 = __builtin_amdgcn_wmma_f32_16x16x32_f16(false, a0, false, b,
                                                      (short)0, acc0, false, false);
        acc1 = __builtin_amdgcn_wmma_f32_16x16x32_f16(false, a1, false, b,
                                                      (short)0, acc1, false, false);
    }

    // ---- fused epilogue: bias (+resid) + activation ----
    float* Yi = Y + (size_t)img * M * P;
    const float* Ri = resid ? resid + (size_t)img * M * P : nullptr;
    const int p = pbase + wn * 16 + (lane & 15);
    #pragma unroll
    for (int r = 0; r < 8; ++r) {
        const int m0 = Mblock + wm * 32 + r + ((lane & 16) ? 8 : 0);
        #pragma unroll
        for (int h = 0; h < 2; ++h) {
            const int m = m0 + h * 16;
            if (m < M) {
                float v = (h ? acc1[r] : acc0[r]) + bias[m];
                if (Ri) v += Ri[(size_t)m * P + p];
                if (act == ACT_RELU)       v = fmaxf(v, 0.f);
                else if (act == ACT_LRELU) v = (v >= 0.f) ? v : 0.1f * v;
                Yi[(size_t)m * P + p] = v;
            }
        }
    }
}

// ---------------------------------------------------------------------------
// Flow-guided deformable conv as WMMA GEMM with gathered B operand.
// Block tile: all 64 out-channels x 64 pixels. 72 K-chunks of 32; chunk kc ->
// tap t = kc/8, deform group d = kc&7 (channels d*32..d*32+31). Sampling
// coords / mask-scaled bilinear weights computed once per pixel per chunk
// into LDS, reused by all 2048 gathered B elements of the chunk.
// grid = (64, 1, 2), block = 256.
// ---------------------------------------------------------------------------
__global__ __launch_bounds__(256)
void dcn_wmma(const float* __restrict__ L1, const float* __restrict__ om,
              const float* __restrict__ flows, const float* __restrict__ Wd,
              const float* __restrict__ bias, float* __restrict__ out, int frame)
{
    __shared__ alignas(32) _Float16 As[4][512];
    __shared__ alignas(32) _Float16 Bs[4][512];
    __shared__ int4   sOff[64];   // clipped corner offsets (y*64+x)
    __shared__ float4 sWt[64];    // mask-scaled bilinear weights (0 if invalid)

    const int b     = blockIdx.z;
    const int pbase = blockIdx.x * 64;
    const float* Xi  = L1 + (size_t)(b * 8 + frame) * 256 * 4096;
    const float* omi = om + (size_t)b * 216 * 4096;
    const float* fl  = (frame == 0) ? nullptr
                                    : flows + (size_t)(b * 7 + frame - 1) * 2 * 4096;
    const int tid  = threadIdx.x, lane = tid & 31, wave = tid >> 5;
    const int wm   = wave >> 2, wn = wave & 3;

    v8f acc0 = {};
    v8f acc1 = {};
    for (int kc = 0; kc < 72; ++kc) {
        const int t  = kc >> 3;
        const int d  = kc & 7;
        const int c0 = d << 5;
        const int ky = t / 3, kx = t - ky * 3;

        __syncthreads();             // protect LDS from previous iteration reads
        if (tid < 64) {
            const int n = tid;
            const int p = pbase + n;
            const int y = p >> 6, x = p & 63;
            const float f0 = fl ? fl[p]        : 0.f;   // flow x
            const float f1 = fl ? fl[4096 + p] : 0.f;   // flow y
            const int oc = 2 * (d * 9 + t);
            const float off0 = omi[(size_t)oc * 4096 + p]       + f1; // y offset
            const float off1 = omi[(size_t)(oc + 1) * 4096 + p] + f0; // x offset
            const float mraw = omi[(size_t)(144 + d * 9 + t) * 4096 + p];
            const float mk   = 1.f / (1.f + expf(-mraw));             // sigmoid
            const float sy = (float)(y + ky - 1) + off0;
            const float sx = (float)(x + kx - 1) + off1;
            const float y0f = floorf(sy), x0f = floorf(sx);
            const int   y0  = (int)y0f,  x0  = (int)x0f;
            const float wy  = sy - y0f,  wx  = sx - x0f;
            const bool vy0 = (y0 >= 0) && (y0 < 64);
            const bool vy1 = (y0 >= -1) && (y0 < 63);
            const bool vx0 = (x0 >= 0) && (x0 < 64);
            const bool vx1 = (x0 >= -1) && (x0 < 63);
            const int y0c = min(max(y0, 0), 63), y1c = min(max(y0 + 1, 0), 63);
            const int x0c = min(max(x0, 0), 63), x1c = min(max(x0 + 1, 0), 63);
            sOff[n] = make_int4(y0c * 64 + x0c, y0c * 64 + x1c,
                                y1c * 64 + x0c, y1c * 64 + x1c);
            sWt[n] = make_float4((vy0 && vx0) ? (1.f - wy) * (1.f - wx) * mk : 0.f,
                                 (vy0 && vx1) ? (1.f - wy) * wx        * mk : 0.f,
                                 (vy1 && vx0) ? wy        * (1.f - wx) * mk : 0.f,
                                 (vy1 && vx1) ? wy        * wx        * mk : 0.f);
        }
        __syncthreads();             // sampling params visible
        #pragma unroll
        for (int j = 0; j < 8; ++j) {
            const int g    = tid * 8 + j;
            const int frag = g >> 9;
            const int L    = (g >> 4) & 31;
            const int e    = g & 15;
            // A: Wt[o, d, c, t] = w_dcn[o][c0+c][ky][kx]
            {
                const int m  = frag * 16 + (L & 15);        // 0..63 == full M
                const int kA = kmapA(e, L);
                As[frag][(L << 4) + e] =
                    (_Float16)Wd[((size_t)m * 256 + (c0 + kA)) * 9 + t];
            }
            // B: bilinear-gathered, mask-scaled samples
            {
                const int n  = frag * 16 + (L & 15);        // pixel in tile
                const int kB = kmapB(e, L);
                const float* base = Xi + (size_t)(c0 + kB) * 4096;
                const int4   o = sOff[n];
                const float4 w = sWt[n];
                const float v = w.x * base[o.x] + w.y * base[o.y] +
                                w.z * base[o.z] + w.w * base[o.w];
                Bs[frag][(L << 4) + e] = (_Float16)v;
            }
        }
        __syncthreads();             // staging visible
        v16h a0 = *reinterpret_cast<const v16h*>(&As[wm * 2 + 0][lane << 4]);
        v16h a1 = *reinterpret_cast<const v16h*>(&As[wm * 2 + 1][lane << 4]);
        v16h bf = *reinterpret_cast<const v16h*>(&Bs[wn][lane << 4]);
        acc0 = __builtin_amdgcn_wmma_f32_16x16x32_f16(false, a0, false, bf,
                                                      (short)0, acc0, false, false);
        acc1 = __builtin_amdgcn_wmma_f32_16x16x32_f16(false, a1, false, bf,
                                                      (short)0, acc1, false, false);
    }

    // bias + leaky relu; out shape (B, N, 64, 64, 64)
    float* Yo = out + (size_t)(b * 8 + frame) * 64 * 4096;
    const int p = pbase + wn * 16 + (lane & 15);
    #pragma unroll
    for (int r = 0; r < 8; ++r) {
        const int m0 = wm * 32 + r + ((lane & 16) ? 8 : 0);
        #pragma unroll
        for (int h = 0; h < 2; ++h) {
            const int m = m0 + h * 16;
            float v = (h ? acc1[r] : acc0[r]) + bias[m];
            v = (v >= 0.f) ? v : 0.1f * v;
            Yo[(size_t)m * 4096 + p] = v;
        }
    }
}

// ---------------------------------------------------------------------------
// Bilinear sample with zero padding outside the 64x64 image.
// ---------------------------------------------------------------------------
__device__ __forceinline__ float bilin64(const float* __restrict__ img,
                                         float sy, float sx)
{
    const float y0f = floorf(sy), x0f = floorf(sx);
    const int   y0  = (int)y0f,  x0  = (int)x0f;
    const float wy  = sy - y0f,  wx  = sx - x0f;
    const int y1 = y0 + 1, x1 = x0 + 1;
    const bool vy0 = (y0 >= 0) && (y0 < 64), vy1 = (y1 >= 0) && (y1 < 64);
    const bool vx0 = (x0 >= 0) && (x0 < 64), vx1 = (x1 >= 0) && (x1 < 64);
    const int y0c = min(max(y0, 0), 63), y1c = min(max(y1, 0), 63);
    const int x0c = min(max(x0, 0), 63), x1c = min(max(x1, 0), 63);
    float a = 0.f;
    if (vy0 && vx0) a += (1.f - wy) * (1.f - wx) * img[y0c * 64 + x0c];
    if (vy0 && vx1) a += (1.f - wy) * wx        * img[y0c * 64 + x1c];
    if (vy1 && vx0) a += wy        * (1.f - wx) * img[y1c * 64 + x0c];
    if (vy1 && vx1) a += wy        * wx        * img[y1c * 64 + x1c];
    return a;
}

// Build concat([flow_warp(nbr, fl), ref, fl]) : (2, 514, 64, 64)
__global__ void warp_concat(const float* __restrict__ L1,
                            const float* __restrict__ flows,
                            float* __restrict__ catb, int frame)
{
    const int idx = blockIdx.x * blockDim.x + threadIdx.x;
    const int tot = 2 * 514 * 4096;
    if (idx >= tot) return;
    const int b   = idx / (514 * 4096);
    const int rem = idx - b * 514 * 4096;
    const int ch  = rem >> 12;
    const int p   = rem & 4095;
    const float* nbr = L1 + (size_t)(b * 8 + frame) * 256 * 4096;
    const float* ref = L1 + (size_t)(b * 8) * 256 * 4096;
    const float* fl  = (frame == 0) ? nullptr
                                    : flows + (size_t)(b * 7 + frame - 1) * 2 * 4096;
    const float f0 = fl ? fl[p]        : 0.f;
    const float f1 = fl ? fl[4096 + p] : 0.f;
    float v;
    if (ch < 256) {
        const int y = p >> 6, x = p & 63;
        v = bilin64(nbr + (size_t)ch * 4096, (float)y + f1, (float)x + f0);
    } else if (ch < 512) {
        v = ref[(size_t)(ch - 256) * 4096 + p];
    } else {
        v = (ch == 512) ? f0 : f1;
    }
    catb[idx] = v;
}

// (16, 256, 64, 64) -> (16, 64, 128, 128), r = 2
__global__ void pixel_shuffle2(const float* __restrict__ in, float* __restrict__ outp)
{
    const int idx = blockIdx.x * blockDim.x + threadIdx.x;
    const int tot = 16 * 64 * 128 * 128;
    if (idx >= tot) return;
    const int img = idx / (64 * 128 * 128);
    const int rem = idx - img * 64 * 128 * 128;
    const int c   = rem / (128 * 128);
    const int yx  = rem - c * 128 * 128;
    const int y   = yx >> 7, x = yx & 127;
    const int ic  = c * 4 + (y & 1) * 2 + (x & 1);
    outp[idx] = in[(((size_t)img * 256 + ic) * 64 + (y >> 1)) * 64 + (x >> 1)];
}

// ---------------------------------------------------------------------------
extern "C" void kernel_launch(void* const* d_in, const int* in_sizes, int n_in,
                              void* d_out, int out_size, void* d_ws, size_t ws_size,
                              hipStream_t stream)
{
    (void)in_sizes; (void)n_in; (void)out_size; (void)ws_size;
    const float* x        = (const float*)d_in[0];   // (2,8,4,64,64)
    const float* flows    = (const float*)d_in[1];   // (2,7,2,64,64)
    const float* w_cf     = (const float*)d_in[2];   const float* b_cf  = (const float*)d_in[3];
    const float* fe_w1    = (const float*)d_in[4];   const float* fe_b1 = (const float*)d_in[5];
    const float* fe_w2    = (const float*)d_in[6];   const float* fe_b2 = (const float*)d_in[7];
    const float* fe_w3    = (const float*)d_in[8];   const float* fe_b3 = (const float*)d_in[9];
    const float* w_after  = (const float*)d_in[10];  const float* b_after = (const float*)d_in[11];
    const float* w_top    = (const float*)d_in[12];  const float* b_top   = (const float*)d_in[13];
    const float* w_off1   = (const float*)d_in[14];  const float* b_off1  = (const float*)d_in[15];
    const float* w_off2   = (const float*)d_in[16];  const float* b_off2  = (const float*)d_in[17];
    const float* w_om     = (const float*)d_in[18];  const float* b_om    = (const float*)d_in[19];
    const float* w_dcn    = (const float*)d_in[20];  const float* b_dcn   = (const float*)d_in[21];
    float* out = (float*)d_out;

    // ---- workspace arena (floats); total 65,798,144 floats ~= 264 MB ----
    float* ws    = (float*)d_ws;
    float* f     = ws;                       //  6,291,456  (16 x  96 x 4096)
    float* big   = f   + 6291456;            // 37,748,736  region, aliased below
    float* r2    = big + 37748736;           //  4,980,736  (16 x  76 x 4096)
    float* L1b   = r2  + 4980736;            // 16,777,216  (16 x  64 x 128 x 128)
    float* r1    = big;                      // 16 x 576 x 4096 (fe loop)
    float* after = big;                      // 16 x 256 x 4096 (post-fe)
    float* ps    = big + 16777216;           // 16 x  64 x 128 x 128
    float* of1   = big;                      //  2 x 256 x 4096 (per-frame)
    float* of2   = big + 2097152;            //  2 x 256 x 4096
    float* omb   = big + 4194304;            //  2 x 216 x 4096
    float* catb  = r2;                       //  2 x 514 x 4096 (fits in r2 region)

    const dim3 blk(256);

    // conv_first: 4 -> 96, 3x3 pad 1, lrelu
    conv_wmma<<<dim3(64, 2, 16), blk, 0, stream>>>(
        x, w_cf, b_cf, nullptr, f, 96, 4, 9, 1, 64, 6, ACT_LRELU);

    // 5 residual feature-extraction blocks
    for (int i = 0; i < 5; ++i) {
        conv_wmma<<<dim3(64, 9, 16), blk, 0, stream>>>(
            f, fe_w1 + (size_t)i * 576 * 96, fe_b1 + i * 576, nullptr,
            r1, 576, 96, 1, 0, 64, 6, ACT_RELU);
        conv_wmma<<<dim3(64, 2, 16), blk, 0, stream>>>(
            r1, fe_w2 + (size_t)i * 76 * 576, fe_b2 + i * 76, nullptr,
            r2, 76, 576, 1, 0, 64, 6, ACT_NONE);
        conv_wmma<<<dim3(64, 2, 16), blk, 0, stream>>>(
            r2, fe_w3 + (size_t)i * 96 * 76 * 9, fe_b3 + i * 96, f /*resid*/,
            f, 96, 76, 9, 1, 64, 6, ACT_NONE);
    }

    // w_after: 96 -> 256, 3x3 pad 1; pixel shuffle x2; w_top 1x1 + lrelu
    conv_wmma<<<dim3(64, 4, 16), blk, 0, stream>>>(
        f, w_after, b_after, nullptr, after, 256, 96, 9, 1, 64, 6, ACT_NONE);
    pixel_shuffle2<<<(16 * 64 * 128 * 128 + 255) / 256, 256, 0, stream>>>(after, ps);
    conv_wmma<<<dim3(256, 1, 16), blk, 0, stream>>>(
        ps, w_top, b_top, nullptr, L1b, 64, 64, 1, 0, 128, 7, ACT_LRELU);
    // L1b per image (64,128,128) reinterpreted as (256,64,64)

    // per-frame alignment
    for (int i = 0; i < 8; ++i) {
        warp_concat<<<(2 * 514 * 4096 + 255) / 256, 256, 0, stream>>>(
            L1b, flows, catb, i);
        conv_wmma<<<dim3(64, 4, 2), blk, 0, stream>>>(
            catb, w_off1, b_off1, nullptr, of1, 256, 514, 9, 1, 64, 6, ACT_LRELU);
        conv_wmma<<<dim3(64, 4, 2), blk, 0, stream>>>(
            of1, w_off2, b_off2, nullptr, of2, 256, 256, 9, 1, 64, 6, ACT_LRELU);
        conv_wmma<<<dim3(64, 4, 2), blk, 0, stream>>>(
            of2, w_om, b_om, nullptr, omb, 216, 256, 9, 1, 64, 6, ACT_NONE);
        dcn_wmma<<<dim3(64, 1, 2), blk, 0, stream>>>(
            L1b, omb, flows, w_dcn, b_dcn, out, i);
    }
}
